// SceneGraphGAT_31482110279997
// MI455X (gfx1250) — compile-verified, matrix operand order checked by reading, not run on hardware
//
#include <hip/hip_runtime.h>
#include <hip/hip_bf16.h>
#include <cstdint>

// ---------------------------------------------------------------------------
// 2-layer GAT for MI455X (gfx1250, wave32).
// GEMMs run on v_wmma_f32_16x16x32_bf16 (bf16 in, f32 accum).
// Edge phase: segment softmax via int-encoded float atomicMax + atomicAdd,
// message scatter with one wave per (edge, head), float4 per lane.
// ---------------------------------------------------------------------------

typedef __attribute__((ext_vector_type(16))) __bf16 v16bf;
typedef __attribute__((ext_vector_type(8)))  __bf16 v8bf;
typedef __attribute__((ext_vector_type(8)))  float  v8f;

union Frag16 { v16bf v; struct { v8bf lo; v8bf hi; } p; };

#define K_DIM 256          // reduction dim for both layers (IN_C == 2*HID)
#define NEG_SLOPE 0.2f

static __device__ __forceinline__ int f2ord(float f) {
    int i = __float_as_int(f);
    return i >= 0 ? i : (i ^ 0x7FFFFFFF);     // monotonic float->int map
}
static __device__ __forceinline__ float ord2f(int i) {
    return __int_as_float(i >= 0 ? i : (i ^ 0x7FFFFFFF));
}

// ----------------------------- small utilities -----------------------------

__global__ void fill_f32_kernel(float* p, float v, long long n) {
    long long i = (long long)blockIdx.x * blockDim.x + threadIdx.x;
    if (i < n) p[i] = v;
}
__global__ void fill_i32_kernel(int* p, int v, long long n) {
    long long i = (long long)blockIdx.x * blockDim.x + threadIdx.x;
    if (i < n) p[i] = v;
}
__global__ void f32_to_bf16_kernel(const float* __restrict__ in,
                                   __bf16* __restrict__ out, long long n) {
    long long i = (long long)blockIdx.x * blockDim.x + threadIdx.x;
    if (i < n) out[i] = (__bf16)in[i];
}
// WT[c*K + k] = W[k*NC + c]   (column-major bf16 copy of W for the B operand)
__global__ void transpose_bf16_kernel(const float* __restrict__ W,
                                      __bf16* __restrict__ WT, int K, int NC) {
    int i = blockIdx.x * blockDim.x + threadIdx.x;
    if (i >= K * NC) return;
    int c = i / K, k = i - c * K;
    WT[i] = (__bf16)W[(long long)k * NC + c];
}

// ------------------------------- WMMA GEMM ---------------------------------
// D[M][NC] = A[M][K_DIM] * BT[NC][K_DIM]^T  (bf16 inputs, f32 accumulate)
// One wave per 16-row tile x 64-column strip (4 N-tiles, A fragment reused 4x).
__global__ __launch_bounds__(256) void gemm_wmma_bf16_kernel(
    const __bf16* __restrict__ A, const __bf16* __restrict__ BT,
    float* __restrict__ D, int M, int NC)
{
    const int lane = threadIdx.x & 31;
    const int wave = threadIdx.x >> 5;
    const int mt   = blockIdx.x * 8 + wave;      // 16-row tile index
    if (mt * 16 >= M) return;                    // wave-uniform exit
    const int ncol0 = blockIdx.y * 64;

    const int half = lane >> 4;                  // 0: lanes 0-15, 1: 16-31
    const int l15  = lane & 15;
    int rowA = mt * 16 + l15;
    if (rowA >= M) rowA = M - 1;                 // clamp (M % 16 == 0 here)

    const __bf16* aRow = A + (size_t)rowA * K_DIM;
    const __bf16* bCol = BT + (size_t)(ncol0 + l15) * K_DIM;

    v8f acc0 = {}, acc1 = {}, acc2 = {}, acc3 = {};

    #pragma unroll
    for (int k0 = 0; k0 < K_DIM; k0 += 32) {
        const int kb = k0 + half * 8;
        Frag16 a, b0, b1, b2, b3;
        // A frag: lanes 0-15 hold K = kb..kb+7 and kb+16..kb+23 of row M=l15
        a.p.lo  = *reinterpret_cast<const v8bf*>(aRow + kb);
        a.p.hi  = *reinterpret_cast<const v8bf*>(aRow + kb + 16);
        b0.p.lo = *reinterpret_cast<const v8bf*>(bCol + kb);
        b0.p.hi = *reinterpret_cast<const v8bf*>(bCol + kb + 16);
        b1.p.lo = *reinterpret_cast<const v8bf*>(bCol + 16 * K_DIM + kb);
        b1.p.hi = *reinterpret_cast<const v8bf*>(bCol + 16 * K_DIM + kb + 16);
        b2.p.lo = *reinterpret_cast<const v8bf*>(bCol + 32 * K_DIM + kb);
        b2.p.hi = *reinterpret_cast<const v8bf*>(bCol + 32 * K_DIM + kb + 16);
        b3.p.lo = *reinterpret_cast<const v8bf*>(bCol + 48 * K_DIM + kb);
        b3.p.hi = *reinterpret_cast<const v8bf*>(bCol + 48 * K_DIM + kb + 16);

        acc0 = __builtin_amdgcn_wmma_f32_16x16x32_bf16(false, a.v, false, b0.v,
                                                       (short)0, acc0, false, false);
        acc1 = __builtin_amdgcn_wmma_f32_16x16x32_bf16(false, a.v, false, b1.v,
                                                       (short)0, acc1, false, false);
        acc2 = __builtin_amdgcn_wmma_f32_16x16x32_bf16(false, a.v, false, b2.v,
                                                       (short)0, acc2, false, false);
        acc3 = __builtin_amdgcn_wmma_f32_16x16x32_bf16(false, a.v, false, b3.v,
                                                       (short)0, acc3, false, false);
    }

    // D layout: VGPR r -> row = mt*16 + half*8 + r, col = ncol0 + 16*t + l15
    float* dBase = D + (size_t)(mt * 16 + half * 8) * NC + ncol0 + l15;
    #pragma unroll
    for (int r = 0; r < 8; ++r) {
        int row = mt * 16 + half * 8 + r;
        if (row < M) {
            dBase[(size_t)r * NC + 0]  = acc0[r];
            dBase[(size_t)r * NC + 16] = acc1[r];
            dBase[(size_t)r * NC + 32] = acc2[r];
            dBase[(size_t)r * NC + 48] = acc3[r];
        }
    }
}

// ----------------------- per-node attention dot products --------------------
// as[n*H+h] = dot(Hf[n, h*C:(h+1)*C], a_s[h]);  ad likewise
__global__ void alpha_kernel(const float* __restrict__ Hf,
                             const float* __restrict__ a_s,
                             const float* __restrict__ a_d,
                             float* __restrict__ as, float* __restrict__ ad,
                             int N, int H, int C)
{
    int idx = blockIdx.x * blockDim.x + threadIdx.x;
    if (idx >= N * H) return;
    int n = idx / H, h = idx - n * H;
    const float* row = Hf + (size_t)n * H * C + (size_t)h * C;
    const float* vs = a_s + (size_t)h * C;
    const float* vd = a_d + (size_t)h * C;
    float s = 0.f, d = 0.f;
    for (int c = 0; c < C; ++c) { float v = row[c]; s += v * vs[c]; d += v * vd[c]; }
    as[idx] = s; ad[idx] = d;
}

// ----------------------------- edge phase ----------------------------------
// pass 1: e = leakyrelu(as[src]+ad[dst]); segment max via atomicMax(int-enc)
__global__ void edge_score_kernel(const int* __restrict__ src,
                                  const int* __restrict__ dst,
                                  const float* __restrict__ as,
                                  const float* __restrict__ ad,
                                  float* __restrict__ ebuf, int* __restrict__ menc,
                                  int E, int Etot, int H)
{
    int idx = blockIdx.x * blockDim.x + threadIdx.x;
    if (idx >= Etot * H) return;
    int e = idx / H, h = idx - e * H;
    int s, d;
    if (e < E) { s = src[e]; d = dst[e]; } else { s = d = e - E; }
    float v = as[s * H + h] + ad[d * H + h];
    v = v > 0.f ? v : NEG_SLOPE * v;
    ebuf[idx] = v;
    atomicMax(&menc[d * H + h], f2ord(v));
}

// pass 2: ex = exp(e - m[dst]); segment sum via atomicAdd
__global__ void edge_exp_kernel(const int* __restrict__ src,
                                const int* __restrict__ dst,
                                float* __restrict__ ebuf,
                                const int* __restrict__ menc,
                                float* __restrict__ denom,
                                int E, int Etot, int H)
{
    int idx = blockIdx.x * blockDim.x + threadIdx.x;
    if (idx >= Etot * H) return;
    int e = idx / H, h = idx - e * H;
    int d = (e < E) ? dst[e] : (e - E);
    float ex = __expf(ebuf[idx] - ord2f(menc[d * H + h]));
    ebuf[idx] = ex;
    atomicAdd(&denom[d * H + h], ex);
}

// pass 3: one wave per (edge, head); agg[dst] += h[src] * (ex/denom[dst])
// C == 128: 32 lanes x float4 covers the head exactly.
__global__ __launch_bounds__(256) void edge_agg_kernel(
    const int* __restrict__ src, const int* __restrict__ dst,
    const float* __restrict__ ebuf, const float* __restrict__ denom,
    const float* __restrict__ Hf, float* __restrict__ agg,
    int E, int Etot, int H, int C)
{
    int gw   = (blockIdx.x * blockDim.x + threadIdx.x) >> 5;
    int lane = threadIdx.x & 31;
    if (gw >= Etot * H) return;
    int e = gw / H, h = gw - e * H;
    int s, d;
    if (e < E) { s = src[e]; d = dst[e]; } else { s = d = e - E; }
    float alpha = ebuf[gw] / denom[d * H + h];

    const float4* hs = reinterpret_cast<const float4*>(
        Hf + (size_t)s * H * C + (size_t)h * C);
    float* ao = agg + (size_t)d * H * C + (size_t)h * C;
    for (int c4 = lane; c4 * 4 < C; c4 += 32) {
        float4 v = hs[c4];
        int c = c4 * 4;
        atomicAdd(ao + c + 0, v.x * alpha);
        atomicAdd(ao + c + 1, v.y * alpha);
        atomicAdd(ao + c + 2, v.z * alpha);
        atomicAdd(ao + c + 3, v.w * alpha);
    }
}

// layer-1 epilogue: (+bias, ELU) -> bf16 input for layer-2 GEMM
__global__ void post_elu_bf16_kernel(const float* __restrict__ agg,
                                     const float* __restrict__ bias,
                                     __bf16* __restrict__ out, long long n, int HC)
{
    long long i = (long long)blockIdx.x * blockDim.x + threadIdx.x;
    if (i >= n) return;
    int c = (int)(i % HC);
    float v = agg[i] + bias[c];
    v = v > 0.f ? v : (__expf(v) - 1.f);
    out[i] = (__bf16)v;
}

// final epilogue: d_out = agg2 + b2
__global__ void final_bias_kernel(const float* __restrict__ agg,
                                  const float* __restrict__ bias,
                                  float* __restrict__ out, long long n, int HC)
{
    long long i = (long long)blockIdx.x * blockDim.x + threadIdx.x;
    if (i >= n) return;
    out[i] = agg[i] + bias[(int)(i % HC)];
}

// ---------------------------------------------------------------------------

static inline long long cdivll(long long a, long long b) { return (a + b - 1) / b; }

extern "C" void kernel_launch(void* const* d_in, const int* in_sizes, int n_in,
                              void* d_out, int out_size, void* d_ws, size_t ws_size,
                              hipStream_t stream) {
    const int IN_C = 256, HID = 128, OUT_C = 128;
    const int HC1 = 2 * HID;                  // 256
    const int N = in_sizes[0] / IN_C;         // 50000
    const int E = in_sizes[1] / 2;            // 800000
    const int Etot = E + N;                   // self-loops appended

    const float* x      = (const float*)d_in[0];
    const int*   src    = (const int*)d_in[1];          // edge_index[0]
    const int*   dst    = src + E;                      // edge_index[1]
    const float* W1     = (const float*)d_in[2];
    const float* a_src1 = (const float*)d_in[3];
    const float* a_dst1 = (const float*)d_in[4];
    const float* b1     = (const float*)d_in[5];
    const float* W2     = (const float*)d_in[6];
    const float* a_src2 = (const float*)d_in[7];
    const float* a_dst2 = (const float*)d_in[8];
    const float* b2     = (const float*)d_in[9];
    float* out = (float*)d_out;

    // ---- carve workspace (256B-aligned slabs) ----
    size_t off = 0;
    auto carve = [&](size_t bytes) -> char* {
        char* p = (char*)d_ws + off;
        off = (off + bytes + 255) & ~(size_t)255;
        return p;
    };
    __bf16* Xbf   = (__bf16*)carve((size_t)N * IN_C * 2);
    __bf16* W1T   = (__bf16*)carve((size_t)HC1 * K_DIM * 2);
    __bf16* W2T   = (__bf16*)carve((size_t)OUT_C * K_DIM * 2);
    float*  Hf1   = (float*) carve((size_t)N * HC1 * 4);
    float*  as1   = (float*) carve((size_t)N * 2 * 4);
    float*  ad1   = (float*) carve((size_t)N * 2 * 4);
    int*    m1    = (int*)   carve((size_t)N * 2 * 4);
    float*  den1  = (float*) carve((size_t)N * 2 * 4);
    float*  e1    = (float*) carve((size_t)Etot * 2 * 4);
    float*  agg1  = (float*) carve((size_t)N * HC1 * 4);
    __bf16* Hbf   = (__bf16*)carve((size_t)N * HC1 * 2);
    float*  Hf2   = (float*) carve((size_t)N * OUT_C * 4);
    float*  as2   = (float*) carve((size_t)N * 4);
    float*  ad2   = (float*) carve((size_t)N * 4);
    int*    m2    = (int*)   carve((size_t)N * 4);
    float*  den2  = (float*) carve((size_t)N * 4);
    float*  e2    = (float*) carve((size_t)Etot * 4);
    float*  agg2  = (float*) carve((size_t)N * OUT_C * 4);

    const int B = 256;
    const int NEG_INF_ENC = (int)(0xFF800000u ^ 0x7FFFFFFFu);  // f2ord(-inf)

    // ---- prep: convert x, transpose+convert weights ----
    {
        long long n = (long long)N * IN_C;
        f32_to_bf16_kernel<<<cdivll(n, B), B, 0, stream>>>(x, Xbf, n);
        transpose_bf16_kernel<<<cdivll((long long)K_DIM * HC1, B), B, 0, stream>>>(W1, W1T, K_DIM, HC1);
        transpose_bf16_kernel<<<cdivll((long long)K_DIM * OUT_C, B), B, 0, stream>>>(W2, W2T, K_DIM, OUT_C);
    }

    const int mtiles = (N + 15) / 16;
    dim3 gemmBlk(256);

    // =========================== layer 1 (H=2, C=128) ===========================
    {
        dim3 gemmGrid((unsigned)cdivll(mtiles, 8), HC1 / 64);
        gemm_wmma_bf16_kernel<<<gemmGrid, gemmBlk, 0, stream>>>(Xbf, W1T, Hf1, N, HC1);

        alpha_kernel<<<cdivll((long long)N * 2, B), B, 0, stream>>>(
            Hf1, a_src1, a_dst1, as1, ad1, N, 2, HID);

        fill_i32_kernel<<<cdivll((long long)N * 2, B), B, 0, stream>>>(m1, NEG_INF_ENC, (long long)N * 2);
        fill_f32_kernel<<<cdivll((long long)N * 2, B), B, 0, stream>>>(den1, 0.f, (long long)N * 2);
        fill_f32_kernel<<<cdivll((long long)N * HC1, B), B, 0, stream>>>(agg1, 0.f, (long long)N * HC1);

        long long ne = (long long)Etot * 2;
        edge_score_kernel<<<cdivll(ne, B), B, 0, stream>>>(src, dst, as1, ad1, e1, m1, E, Etot, 2);
        edge_exp_kernel<<<cdivll(ne, B), B, 0, stream>>>(src, dst, e1, m1, den1, E, Etot, 2);
        edge_agg_kernel<<<cdivll(ne * 32, B), B, 0, stream>>>(src, dst, e1, den1, Hf1, agg1, E, Etot, 2, HID);

        long long n = (long long)N * HC1;
        post_elu_bf16_kernel<<<cdivll(n, B), B, 0, stream>>>(agg1, b1, Hbf, n, HC1);
    }

    // =========================== layer 2 (H=1, C=128) ===========================
    {
        dim3 gemmGrid((unsigned)cdivll(mtiles, 8), OUT_C / 64);
        gemm_wmma_bf16_kernel<<<gemmGrid, gemmBlk, 0, stream>>>(Hbf, W2T, Hf2, N, OUT_C);

        alpha_kernel<<<cdivll((long long)N, B), B, 0, stream>>>(
            Hf2, a_src2, a_dst2, as2, ad2, N, 1, OUT_C);

        fill_i32_kernel<<<cdivll((long long)N, B), B, 0, stream>>>(m2, NEG_INF_ENC, (long long)N);
        fill_f32_kernel<<<cdivll((long long)N, B), B, 0, stream>>>(den2, 0.f, (long long)N);
        fill_f32_kernel<<<cdivll((long long)N * OUT_C, B), B, 0, stream>>>(agg2, 0.f, (long long)N * OUT_C);

        long long ne = (long long)Etot;
        edge_score_kernel<<<cdivll(ne, B), B, 0, stream>>>(src, dst, as2, ad2, e2, m2, E, Etot, 1);
        edge_exp_kernel<<<cdivll(ne, B), B, 0, stream>>>(src, dst, e2, m2, den2, E, Etot, 1);
        edge_agg_kernel<<<cdivll(ne * 32, B), B, 0, stream>>>(src, dst, e2, den2, Hf2, agg2, E, Etot, 1, OUT_C);

        long long n = (long long)N * OUT_C;
        final_bias_kernel<<<cdivll(n, B), B, 0, stream>>>(agg2, b2, out, n, OUT_C);
    }
    (void)n_in; (void)out_size; (void)ws_size;
}
